// MemVim_57578331570284
// MI455X (gfx1250) — matrix operand order repaired and checked by reading, not compiled, not run on hardware
//
#include <hip/hip_runtime.h>
#include <hip/hip_bf16.h>

typedef __attribute__((ext_vector_type(16))) __bf16 v16bf;
typedef __attribute__((ext_vector_type(8)))  float  v8f;

#define Bb 32
#define Nn 196
#define Dd 1024
#define Kk 512
#define TOK (Bb*Nn)              // 6272
#define DEC_N (7*7*512)          // 25088
#define XRECON (Bb*3*28*28)      // 75264

#define WMMA_BF16(a, b, c) \
  __builtin_amdgcn_wmma_f32_16x16x32_bf16(false, (a), false, (b), (short)0, (c), false, false)

// ---------------------------------------------------------------------------
// Fragment loaders using contiguous k = 16*(lane/16) + s (16-bit A 16x32, B 32x16, f32 C/D)
// ---------------------------------------------------------------------------
__device__ __forceinline__ v16bf load_a_frag(const __bf16* __restrict__ A, int lda,
                                             int lane, int m0, int k0) {
  int m  = m0 + (lane & 15);
  int kb = k0 + ((lane & 16) ? 16 : 0);
  const __bf16* r = A + (size_t)m * lda + kb;
  v16bf f;
#pragma unroll
  for (int i = 0; i < 4; ++i) { f[2*i] = r[2*i];     f[2*i+1] = r[2*i+1]; }
#pragma unroll
  for (int i = 4; i < 8; ++i) { f[2*i] = r[2*i]; f[2*i+1] = r[2*i+1]; }
  return f;
}

// B given N-major (N x K row-major, i.e. B[k][n] = src[n*ldp + k]) -> contiguous
__device__ __forceinline__ v16bf load_bt_frag(const __bf16* __restrict__ P, int ldp,
                                              int lane, int k0, int n0) {
  int n  = n0 + (lane & 15);
  int kb = k0 + ((lane & 16) ? 16 : 0);
  const __bf16* r = P + (size_t)n * ldp + kb;
  v16bf f;
#pragma unroll
  for (int j = 0; j < 16; ++j) f[j] = r[j];
  return f;
}

// ---------------------------------------------------------------------------
// Stage 1: per-row L2 norm + f32 -> bf16 conversion
// ---------------------------------------------------------------------------
__global__ void rownorm_cvt(const float* __restrict__ src, __bf16* __restrict__ dst,
                            float* __restrict__ norms, int D) {
  __shared__ float red[256];
  int t = threadIdx.x;
  size_t base = (size_t)blockIdx.x * D;
  float ss = 0.f;
  for (int i = t; i < D; i += 256) {
    float v = src[base + i];
    dst[base + i] = (__bf16)v;
    ss += v * v;
  }
  red[t] = ss; __syncthreads();
  for (int s = 128; s > 0; s >>= 1) { if (t < s) red[t] += red[t + s]; __syncthreads(); }
  if (t == 0) norms[blockIdx.x] = sqrtf(red[0]);
}

// transpose-convert: src [R x C] f32 row-major -> dst [C x R] bf16 row-major
__global__ void cvt_bf16_T(const float* __restrict__ src, __bf16* __restrict__ dst,
                           int R, int C) {
  int idx = blockIdx.x * blockDim.x + threadIdx.x;
  if (idx >= R * C) return;
  int r = idx / C, c = idx % C;               // coalesced read
  dst[(size_t)c * R + r] = (__bf16)src[idx];
}

// ---------------------------------------------------------------------------
// Stage 2: sim = (z @ P^T) / max(|z||p|, eps)   [TOK x Kk], inner = Dd
// wave tile: 16 x 64 (4 accumulators, shared A frag)
// ---------------------------------------------------------------------------
__global__ void __launch_bounds__(128) gemm_sim(
    const __bf16* __restrict__ zb, const __bf16* __restrict__ pb,
    const float* __restrict__ zn, const float* __restrict__ pn,
    float* __restrict__ sim) {
  int lane = threadIdx.x & 31, wave = threadIdx.x >> 5;
  int m0 = blockIdx.x * 16;
  int n0 = blockIdx.y * 256 + wave * 64;
  v8f acc[4] = {};
  for (int k0 = 0; k0 < Dd; k0 += 32) {
    __builtin_prefetch(zb + (size_t)(m0 + (lane & 15)) * Dd + k0 + 64, 0, 0);
    v16bf a = load_a_frag(zb, Dd, lane, m0, k0);
#pragma unroll
    for (int j = 0; j < 4; ++j) {
      v16bf b = load_bt_frag(pb, Dd, lane, k0, n0 + 16 * j);
      acc[j] = WMMA_BF16(a, b, acc[j]);
    }
  }
  int mrow = m0 + ((lane & 16) ? 8 : 0);
#pragma unroll
  for (int j = 0; j < 4; ++j) {
    int n = n0 + 16 * j + (lane & 15);
    float pnv = pn[n];
#pragma unroll
    for (int i = 0; i < 8; ++i) {
      int m = mrow + i;
      float denom = fmaxf(zn[m] * pnv, 1e-8f);
      sim[(size_t)m * Kk + n] = acc[j][i] * __builtin_amdgcn_rcpf(denom);
    }
  }
}

// ---------------------------------------------------------------------------
// Stage 3: softmax over K=512 per token; write f32 attn (d_out) + bf16 copy
// ---------------------------------------------------------------------------
__global__ void softmax_rows(const float* __restrict__ sim, float* __restrict__ attn,
                             __bf16* __restrict__ attn_bf) {
  __shared__ float red[256];
  int t = threadIdx.x;
  size_t base = (size_t)blockIdx.x * Kk;
  float a0 = sim[base + t], a1 = sim[base + t + 256];
  red[t] = fmaxf(a0, a1); __syncthreads();
  for (int s = 128; s > 0; s >>= 1) { if (t < s) red[t] = fmaxf(red[t], red[t + s]); __syncthreads(); }
  float M = red[0]; __syncthreads();
  float e0 = __expf(a0 - M), e1 = __expf(a1 - M);
  red[t] = e0 + e1; __syncthreads();
  for (int s = 128; s > 0; s >>= 1) { if (t < s) red[t] += red[t + s]; __syncthreads(); }
  float inv = __builtin_amdgcn_rcpf(red[0]);
  float r0 = e0 * inv, r1 = e1 * inv;
  attn[base + t] = r0;            attn[base + t + 256] = r1;
  attn_bf[base + t] = (__bf16)r0; attn_bf[base + t + 256] = (__bf16)r1;
}

// ---------------------------------------------------------------------------
// Stage 4: z_hat = attn @ P   [TOK x Dd], inner = Kk; B from pbT (N-major)
// ---------------------------------------------------------------------------
__global__ void __launch_bounds__(128) gemm_zhat(
    const __bf16* __restrict__ ab, const __bf16* __restrict__ pbT,
    float* __restrict__ zhat, __bf16* __restrict__ zhb) {
  int lane = threadIdx.x & 31, wave = threadIdx.x >> 5;
  int m0 = blockIdx.x * 16;
  int n0 = blockIdx.y * 256 + wave * 64;
  v8f acc[4] = {};
  for (int k0 = 0; k0 < Kk; k0 += 32) {
    v16bf a = load_a_frag(ab, Kk, lane, m0, k0);
#pragma unroll
    for (int j = 0; j < 4; ++j) {
      v16bf b = load_bt_frag(pbT, Kk, lane, k0, n0 + 16 * j);
      acc[j] = WMMA_BF16(a, b, acc[j]);
    }
  }
  int mrow = m0 + ((lane & 16) ? 8 : 0);
#pragma unroll
  for (int j = 0; j < 4; ++j) {
    int n = n0 + 16 * j + (lane & 15);
#pragma unroll
    for (int i = 0; i < 8; ++i) {
      int m = mrow + i;
      float v = acc[j][i];
      zhat[(size_t)m * Dd + n] = v;
      zhb [(size_t)m * Dd + n] = (__bf16)v;
    }
  }
}

// ---------------------------------------------------------------------------
// Stage 5: gate = sigmoid([z|z_hat] @ Wg + bg); z_fused = g*z + (1-g)*z_hat
// inner = 2*Dd. A tile (16x32 bf16 = 1KB) staged in LDS with
// global_load_async_to_lds_b64 (ASYNCcnt), double-buffered, shared by 4 waves.
// ---------------------------------------------------------------------------
__global__ void __launch_bounds__(128) gemm_gate_fuse(
    const __bf16* __restrict__ zb, const __bf16* __restrict__ zhb,
    const __bf16* __restrict__ gwbT, const float* __restrict__ gbias,
    const float* __restrict__ z, const float* __restrict__ zhat,
    float* __restrict__ zfused) {
  extern __shared__ char smem_raw[];
  __bf16* smem = (__bf16*)smem_raw;            // 2 x 512 bf16 (16 rows x 32 k)
  int tid = threadIdx.x;
  int lane = tid & 31, wave = tid >> 5;
  int m0 = blockIdx.x * 16;
  int n0 = blockIdx.y * 256 + wave * 64;
  const int NK = (2 * Dd) / 32;                // 64 k-steps

  auto stage = [&](int step, int buf) {
    int kk = step * 32;
    const __bf16* Ap = (kk < Dd) ? zb : zhb;
    int kx = (kk < Dd) ? kk : (kk - Dd);
    int row  = tid >> 3;                       // 0..15
    int cole = (tid & 7) * 4;                  // element offset (8B per lane)
    const __bf16* gsrc = Ap + (size_t)(m0 + row) * Dd + kx + cole;
    unsigned lds_off = (unsigned)(buf * 1024 + tid * 8);   // byte offset
    asm volatile("global_load_async_to_lds_b64 %0, %1, off"
                 :: "v"(lds_off), "v"(gsrc) : "memory");
  };

  v8f acc[4] = {};
  stage(0, 0);
  asm volatile("s_wait_asynccnt 0x0" ::: "memory");
  __syncthreads();
  for (int i = 0; i < NK; ++i) {
    int buf = i & 1;
    if (i + 1 < NK) stage(i + 1, buf ^ 1);     // overlap with compute below
    int k0 = i * 32;
    v16bf a = load_a_frag(smem + buf * 512, 32, lane, 0, 0);
#pragma unroll
    for (int j = 0; j < 4; ++j) {
      v16bf b = load_bt_frag(gwbT, 2 * Dd, lane, k0, n0 + 16 * j);
      acc[j] = WMMA_BF16(a, b, acc[j]);
    }
    asm volatile("s_wait_asynccnt 0x0" ::: "memory");
    __syncthreads();
  }

  int mrow = m0 + ((lane & 16) ? 8 : 0);
#pragma unroll
  for (int j = 0; j < 4; ++j) {
    int n = n0 + 16 * j + (lane & 15);
    float bg = gbias[n];
#pragma unroll
    for (int i = 0; i < 8; ++i) {
      int m = mrow + i;
      float e = __expf(-(acc[j][i] + bg));
      float g = __builtin_amdgcn_rcpf(1.0f + e);
      size_t o = (size_t)m * Dd + n;
      zfused[o] = g * z[o] + (1.0f - g) * zhat[o];
    }
  }
}

// ---------------------------------------------------------------------------
// Stage 6: pooled[b,d] = mean over 196 tokens
// ---------------------------------------------------------------------------
__global__ void pool_mean(const float* __restrict__ zf, float* __restrict__ pooled) {
  int idx = blockIdx.x * blockDim.x + threadIdx.x;   // 32*1024
  int b = idx >> 10, d = idx & 1023;
  float s = 0.f;
  const float* p = zf + ((size_t)b * Nn) * Dd + d;
  for (int n = 0; n < Nn; ++n) s += p[(size_t)n * Dd];
  pooled[idx] = s * (1.0f / (float)Nn);
}

// ---------------------------------------------------------------------------
// Stage 7: h = pooled @ dec_w + dec_b  (bandwidth-bound: 103MB weights, f32)
// ---------------------------------------------------------------------------
__global__ void dec_fc(const float* __restrict__ pooled, const float* __restrict__ dw,
                       const float* __restrict__ db, float* __restrict__ h1) {
  int idx = blockIdx.x * blockDim.x + threadIdx.x;   // 32*25088
  int b = idx / DEC_N, j = idx % DEC_N;
  float acc = db[j];
  const float* pr = pooled + (size_t)b * Dd;
  for (int d = 0; d < Dd; ++d) acc += pr[d] * dw[(size_t)d * DEC_N + j];
  h1[idx] = acc;    // layout [B,512,7,7] row-major
}

// ---------------------------------------------------------------------------
// Stage 8: ConvTranspose2d(512->256, k4, s2, p1) + ReLU   7x7 -> 14x14
// ---------------------------------------------------------------------------
__global__ void convt1(const float* __restrict__ h1, const float* __restrict__ w,
                       const float* __restrict__ bias, float* __restrict__ h2) {
  int idx = blockIdx.x * blockDim.x + threadIdx.x;   // 32*256*14*14
  int ox = idx % 14, oy = (idx / 14) % 14, oc = (idx / 196) % 256, b = idx / (196 * 256);
  float acc = bias[oc];
  for (int ky = 0; ky < 4; ++ky) {
    int ty = oy + 1 - ky;
    if (ty < 0 || (ty & 1)) continue;
    int iy = ty >> 1; if (iy >= 7) continue;
    for (int kx = 0; kx < 4; ++kx) {
      int tx = ox + 1 - kx;
      if (tx < 0 || (tx & 1)) continue;
      int ix = tx >> 1; if (ix >= 7) continue;
      const float* hp = h1 + (size_t)b * 512 * 49 + iy * 7 + ix;
      const float* wp = w + (size_t)oc * 16 + ky * 4 + kx;
      float s = 0.f;
      for (int ic = 0; ic < 512; ++ic)
        s += hp[(size_t)ic * 49] * wp[(size_t)ic * 256 * 16];
      acc += s;
    }
  }
  h2[idx] = fmaxf(acc, 0.f);   // layout [B,256,14,14]
}

// ---------------------------------------------------------------------------
// Stage 9: ConvTranspose2d(256->3, k4, s2, p1)  14x14 -> 28x28 -> x_recon
// ---------------------------------------------------------------------------
__global__ void convt2(const float* __restrict__ h2, const float* __restrict__ w,
                       const float* __restrict__ bias, float* __restrict__ out) {
  int idx = blockIdx.x * blockDim.x + threadIdx.x;   // 32*3*28*28
  int ox = idx % 28, oy = (idx / 28) % 28, oc = (idx / 784) % 3, b = idx / 2352;
  float acc = bias[oc];
  for (int ky = 0; ky < 4; ++ky) {
    int ty = oy + 1 - ky;
    if (ty < 0 || (ty & 1)) continue;
    int iy = ty >> 1; if (iy >= 14) continue;
    for (int kx = 0; kx < 4; ++kx) {
      int tx = ox + 1 - kx;
      if (tx < 0 || (tx & 1)) continue;
      int ix = tx >> 1; if (ix >= 14) continue;
      const float* hp = h2 + (size_t)b * 256 * 196 + iy * 14 + ix;
      const float* wp = w + (size_t)oc * 16 + ky * 4 + kx;
      float s = 0.f;
      for (int ic = 0; ic < 256; ++ic)
        s += hp[(size_t)ic * 196] * wp[(size_t)ic * 3 * 16];
      acc += s;
    }
  }
  out[idx] = acc;   // x_recon [B,3,28,28]
}

// ---------------------------------------------------------------------------
extern "C" void kernel_launch(void* const* d_in, const int* in_sizes, int n_in,
                              void* d_out, int out_size, void* d_ws, size_t ws_size,
                              hipStream_t stream) {
  const float* z    = (const float*)d_in[0];
  const float* prot = (const float*)d_in[1];
  const float* gw   = (const float*)d_in[2];
  const float* gb   = (const float*)d_in[3];
  const float* dw   = (const float*)d_in[4];
  const float* db   = (const float*)d_in[5];
  const float* w1   = (const float*)d_in[6];
  const float* b1   = (const float*)d_in[7];
  const float* w2   = (const float*)d_in[8];
  const float* b2   = (const float*)d_in[9];

  float* x_recon = (float*)d_out;
  float* attn    = (float*)d_out + XRECON;

  char* ws = (char*)d_ws;
  size_t off = 0;
#define TAKE(T, name, bytes) T name = (T)(ws + off); off += (((size_t)(bytes)) + 255) & ~(size_t)255
  TAKE(__bf16*, zb,     (size_t)TOK * Dd * 2);
  TAKE(__bf16*, pb,     (size_t)Kk * Dd * 2);           // [K x D] row-major
  TAKE(__bf16*, pbT,    (size_t)Dd * Kk * 2);           // [D x K] N-major for GEMM2
  TAKE(float*,  zn,     (size_t)TOK * 4);
  TAKE(float*,  pn,     (size_t)Kk * 4);
  TAKE(float*,  sim,    (size_t)TOK * Kk * 4);
  TAKE(__bf16*, attnb,  (size_t)TOK * Kk * 2);
  TAKE(float*,  zhat,   (size_t)TOK * Dd * 4);
  TAKE(__bf16*, zhb,    (size_t)TOK * Dd * 2);
  TAKE(__bf16*, gwbT,   (size_t)2 * Dd * Dd * 2);       // [D x 2D] N-major for GEMM3
  TAKE(float*,  zfused, (size_t)TOK * Dd * 4);
  TAKE(float*,  pooled, (size_t)Bb * Dd * 4);
  TAKE(float*,  h1,     (size_t)Bb * DEC_N * 4);
  TAKE(float*,  h2,     (size_t)Bb * 256 * 196 * 4);
#undef TAKE

  // 1) convert + norms (+ N-major transposed weights for contiguous B frags)
  rownorm_cvt<<<TOK, 256, 0, stream>>>(z, zb, zn, Dd);
  rownorm_cvt<<<Kk, 256, 0, stream>>>(prot, pb, pn, Dd);
  cvt_bf16_T<<<(Kk * Dd) / 256, 256, 0, stream>>>(prot, pbT, Kk, Dd);
  cvt_bf16_T<<<(2 * Dd * Dd) / 256, 256, 0, stream>>>(gw, gwbT, 2 * Dd, Dd);

  // 2) cosine-sim GEMM (WMMA bf16, 16x64 tiles per wave)
  gemm_sim<<<dim3(TOK / 16, Kk / 256), 128, 0, stream>>>(zb, pb, zn, pn, sim);

  // 3) softmax -> attn (output #2) + bf16 copy
  softmax_rows<<<TOK, 256, 0, stream>>>(sim, attn, attnb);

  // 4) z_hat GEMM (WMMA bf16)
  gemm_zhat<<<dim3(TOK / 16, Dd / 256), 128, 0, stream>>>(attnb, pbT, zhat, zhb);

  // 5) gate GEMM + sigmoid + fuse (WMMA bf16, async-LDS A staging, 2KB LDS)
  gemm_gate_fuse<<<dim3(TOK / 16, Dd / 256), 128, 2048, stream>>>(
      zb, zhb, gwbT, gb, z, zhat, zfused);

  // 6) token-mean pool
  pool_mean<<<(Bb * Dd) / 256, 256, 0, stream>>>(zfused, pooled);

  // 7) decoder FC (bandwidth-bound, f32)
  dec_fc<<<(Bb * DEC_N) / 256, 256, 0, stream>>>(pooled, dw, db, h1);

  // 8) convT1 + ReLU
  convt1<<<(Bb * 256 * 196) / 256, 256, 0, stream>>>(h1, w1, b1, h2);

  // 9) convT2 -> x_recon (output #1)
  convt2<<<(Bb * 3 * 784) / 256, 256, 0, stream>>>(h2, w2, b2, x_recon);
}
